// TensionBlock_60507499266067
// MI455X (gfx1250) — matrix operand order repaired and checked by reading, not compiled, not run on hardware
//
#include <hip/hip_runtime.h>
#include <math.h>

// ---------------------------------------------------------------------------
// Problem constants (match reference)
// ---------------------------------------------------------------------------
#define B_    2
#define T_    4096
#define D_    512
#define H_    8
#define W_    32
#define HD_   64
#define HID_  1536
#define ROWS  (B_ * T_)              // 8192

static constexpr float kScaleInv = 1.0f / 8.0f;   // 1/sqrt(HD)
static constexpr float kEps      = 1e-5f;

typedef __attribute__((ext_vector_type(16))) _Float16 v16h;
typedef __attribute__((ext_vector_type(8)))  _Float16 v8h;
typedef __attribute__((ext_vector_type(8)))  float    v8f;

// ---------------------------------------------------------------------------
// fp32 -> f16 transposed weight conversion:  wt[n*K + k] = (f16) w[k*N + n]
// ---------------------------------------------------------------------------
__global__ __launch_bounds__(256) void transpose_f16_kernel(
    const float* __restrict__ w, _Float16* __restrict__ wt, int K, int N)
{
    size_t i = (size_t)blockIdx.x * blockDim.x + threadIdx.x;
    size_t total = (size_t)K * N;
    if (i >= total) return;
    int n = (int)(i % N);
    int k = (int)(i / N);
    wt[(size_t)n * K + k] = (_Float16)w[i];
}

// ---------------------------------------------------------------------------
// WMMA GEMM:  C[M,N] = A[M,K] (f16, row-major) * Bt[N,K]^T (f16, N-major)
//
// One wave -> one 16x64 C strip (4 N-subtiles, 4 independent accumulators).
// Software-pipelined: the NEXT k-step's A + 4xB fragments are loaded into a
// second register buffer BEFORE the current step's 4 WMMAs issue, so the
// waits ahead of each v_wmma are covered by loads launched a full iteration
// earlier (no s_wait_loadcnt 0 stalls against just-issued loads).
// 8 waves per block stack 8 M-tiles over one 64-wide N chunk -> B stream is
// reused 8x through L0/L2.
// Requires: K % 32 == 0 (and K >= 64), M % 128 == 0, N % 64 == 0.
// ---------------------------------------------------------------------------
__global__ __launch_bounds__(256)
__attribute__((amdgpu_waves_per_eu(1)))
void gemm_f16_wmma(
    const _Float16* __restrict__ A, const _Float16* __restrict__ Bt,
    float* __restrict__ C, int M, int N, int K)
{
    const int lane  = threadIdx.x & 31;
    const int wave  = threadIdx.x >> 5;
    const int tileM = blockIdx.x * 8 + wave;
    const int nBase = blockIdx.y * 64;
    if (tileM * 16 >= M) return;                 // uniform per wave

    const int mn = lane & 15;                    // row (A) / col (B) within tile
    const int hi = lane >> 4;                    // lane-half select

    const _Float16* arow = A  + (size_t)(tileM * 16 + mn) * K;
    const _Float16* brow = Bt + (size_t)(nBase + mn) * K; // N-subtile j at +16*j*K
    const size_t bstr = (size_t)16 * K;

    const int kaoff = hi * 16;   // A: halves0-15 @ k0+16*hi (contiguous 32B)
    const int kboff = hi * 16;   // B: halves0-15 @ k0+16*hi (contiguous 32B)

    auto loadA = [&](int k0) -> v16h {
        v8h lo = *(const v8h*)(arow + k0 + kaoff);
        v8h hi8 = *(const v8h*)(arow + k0 + kaoff + 8);
        return __builtin_shufflevector(lo, hi8,
                 0, 1, 2, 3, 4, 5, 6, 7, 8, 9, 10, 11, 12, 13, 14, 15);
    };
    auto loadB = [&](int k0, int j) -> v16h {
        return *(const v16h*)(brow + (size_t)j * bstr + k0 + kboff);
    };

    v8f acc0 = {}, acc1 = {}, acc2 = {}, acc3 = {};

    // ---- pipeline prologue: stage k0 = 0 ----
    v16h a  = loadA(0);
    v16h b0 = loadB(0, 0);
    v16h b1 = loadB(0, 1);
    v16h b2 = loadB(0, 2);
    v16h b3 = loadB(0, 3);

    // ---- steady state: prefetch k0+32 fragments, then consume k0 ----
    for (int k0 = 0; k0 < K - 32; k0 += 32) {
        const int kn = k0 + 32;
        __builtin_prefetch(arow + kn + 128, 0, 1);   // global_prefetch_b8
        v16h an  = loadA(kn);
        v16h bn0 = loadB(kn, 0);
        v16h bn1 = loadB(kn, 1);
        v16h bn2 = loadB(kn, 2);
        v16h bn3 = loadB(kn, 3);
        acc0 = __builtin_amdgcn_wmma_f32_16x16x32_f16(false, a, false, b0, (short)0, acc0, false, false);
        acc1 = __builtin_amdgcn_wmma_f32_16x16x32_f16(false, a, false, b1, (short)0, acc1, false, false);
        acc2 = __builtin_amdgcn_wmma_f32_16x16x32_f16(false, a, false, b2, (short)0, acc2, false, false);
        acc3 = __builtin_amdgcn_wmma_f32_16x16x32_f16(false, a, false, b3, (short)0, acc3, false, false);
        a = an; b0 = bn0; b1 = bn1; b2 = bn2; b3 = bn3;
    }
    // ---- epilogue: last k-step ----
    acc0 = __builtin_amdgcn_wmma_f32_16x16x32_f16(false, a, false, b0, (short)0, acc0, false, false);
    acc1 = __builtin_amdgcn_wmma_f32_16x16x32_f16(false, a, false, b1, (short)0, acc1, false, false);
    acc2 = __builtin_amdgcn_wmma_f32_16x16x32_f16(false, a, false, b2, (short)0, acc2, false, false);
    acc3 = __builtin_amdgcn_wmma_f32_16x16x32_f16(false, a, false, b3, (short)0, acc3, false, false);

    // C/D layout: reg r, lane l -> row 16*tileM + 8*hi + r, col nBase+16*j+(l&15)
    float* cb = C + (size_t)(tileM * 16 + hi * 8) * N + nBase + mn;
    #pragma unroll
    for (int r = 0; r < 8; ++r) cb[(size_t)r * N]      = acc0[r];
    #pragma unroll
    for (int r = 0; r < 8; ++r) cb[(size_t)r * N + 16] = acc1[r];
    #pragma unroll
    for (int r = 0; r < 8; ++r) cb[(size_t)r * N + 32] = acc2[r];
    #pragma unroll
    for (int r = 0; r < 8; ++r) cb[(size_t)r * N + 48] = acc3[r];
}

// ---------------------------------------------------------------------------
// Fused LayerNorm (+optional residual) (+optional sinusoidal modulation),
// writing fp32 and/or f16 copies. One 256-thread block per row (D=512).
// ---------------------------------------------------------------------------
__global__ __launch_bounds__(256) void ln_kernel(
    const float* __restrict__ x,  const float* __restrict__ res,
    const float* __restrict__ g,  const float* __restrict__ bb,
    float* __restrict__ out32,    _Float16* __restrict__ out16,
    const float* __restrict__ freqs, const float* __restrict__ phases,
    const float* __restrict__ amp)
{
    __shared__ float sSum[256];
    __shared__ float sSq[256];

    const int    row  = blockIdx.x;
    const int    tid  = threadIdx.x;
    const size_t base = (size_t)row * D_;
    const int d0 = tid, d1 = tid + 256;

    float x0 = x[base + d0];
    float x1 = x[base + d1];
    if (res) { x0 += res[base + d0]; x1 += res[base + d1]; }

    sSum[tid] = x0 + x1;
    sSq[tid]  = x0 * x0 + x1 * x1;
    __syncthreads();
    for (int s = 128; s > 0; s >>= 1) {
        if (tid < s) { sSum[tid] += sSum[tid + s]; sSq[tid] += sSq[tid + s]; }
        __syncthreads();
    }
    const float mu   = sSum[0] * (1.0f / D_);
    const float var  = sSq[0] * (1.0f / D_) - mu * mu;
    const float rstd = rsqrtf(var + kEps);

    float y0 = (x0 - mu) * rstd * g[d0] + bb[d0];
    float y1 = (x1 - mu) * rstd * g[d1] + bb[d1];

    if (freqs) {
        const float a = amp[0];
        const float t = (float)(row % T_);
        y0 *= 1.0f + a * __sinf(t * freqs[d0] + phases[d0]);
        y1 *= 1.0f + a * __sinf(t * freqs[d1] + phases[d1]);
    }
    if (out32) { out32[base + d0] = y0;           out32[base + d1] = y1; }
    if (out16) { out16[base + d0] = (_Float16)y0; out16[base + d1] = (_Float16)y1; }
}

// ---------------------------------------------------------------------------
// Sliding-window sigmoid attention. Window = previous 32 tokens (t-1..t-32),
// zero-padded before t=0 (zero-padded v => zero contribution).
// One 64-thread block per (b,t,h): 32 lanes build tau[w] in LDS, all 64
// lanes then accumulate msg[d] = sum_w tau[w] * v[t-1-w][d].
// (~1% of total FLOPs -> VALU fp32 is the right tool; WMMA stays on GEMMs.)
// ---------------------------------------------------------------------------
__global__ __launch_bounds__(64) void attn_kernel(
    const float* __restrict__ q, const float* __restrict__ kv,
    _Float16* __restrict__ msg16)
{
    __shared__ float qs[HD_];
    __shared__ float tau[W_];

    const int idx = blockIdx.x;          // ((b*T)+t)*H + h
    const int h   = idx % H_;
    const int row = idx / H_;            // b*T + t
    const int t   = row % T_;
    const int tid = threadIdx.x;         // 0..63

    const float* qp = q + (size_t)row * D_ + h * HD_;
    qs[tid] = qp[tid];
    __syncthreads();

    if (tid < W_) {
        float s = 0.0f;
        const int tk = t - 1 - tid;
        if (tk >= 0) {
            const float* kp = kv + (size_t)(row - 1 - tid) * (2 * D_) + h * (2 * HD_);
            #pragma unroll 8
            for (int d = 0; d < HD_; ++d) s += qs[d] * kp[d];
            s = 1.0f / (1.0f + __expf(-s * kScaleInv));
        }
        tau[tid] = s;                    // zero where padded (v is zero there too)
    }
    __syncthreads();

    float acc = 0.0f;
    const int wmax = (t < W_) ? t : W_;
    for (int w = 0; w < wmax; ++w) {
        const float* vp = kv + (size_t)(row - 1 - w) * (2 * D_) + h * (2 * HD_) + HD_;
        acc += tau[w] * vp[tid];
    }
    msg16[(size_t)row * D_ + h * HD_ + tid] = (_Float16)acc;
}

// ---------------------------------------------------------------------------
// zact = silu(g) * v  -> f16 (feeds the w_proj WMMA GEMM)
// ---------------------------------------------------------------------------
__global__ __launch_bounds__(256) void silu_mul_kernel(
    const float* __restrict__ g, const float* __restrict__ v,
    _Float16* __restrict__ out, size_t n)
{
    size_t i = (size_t)blockIdx.x * blockDim.x + threadIdx.x;
    if (i >= n) return;
    float gv = g[i];
    float s  = gv / (1.0f + __expf(-gv));
    out[i] = (_Float16)(s * v[i]);
}

// ---------------------------------------------------------------------------
// Host-side orchestration
// ---------------------------------------------------------------------------
extern "C" void kernel_launch(void* const* d_in, const int* in_sizes, int n_in,
                              void* d_out, int out_size, void* d_ws, size_t ws_size,
                              hipStream_t stream)
{
    (void)in_sizes; (void)n_in; (void)out_size; (void)ws_size;

    const float* x      = (const float*)d_in[0];
    const float* pre_g  = (const float*)d_in[1];
    const float* pre_b  = (const float*)d_in[2];
    const float* wq     = (const float*)d_in[3];
    const float* wkv    = (const float*)d_in[4];
    const float* wo     = (const float*)d_in[5];
    const float* attn_g = (const float*)d_in[6];
    const float* attn_b = (const float*)d_in[7];
    const float* freqs  = (const float*)d_in[8];
    const float* phases = (const float*)d_in[9];
    const float* amp    = (const float*)d_in[10];
    const float* w_gate = (const float*)d_in[11];
    const float* w_val  = (const float*)d_in[12];
    const float* w_proj = (const float*)d_in[13];
    const float* ffn_g  = (const float*)d_in[14];
    const float* ffn_b  = (const float*)d_in[15];

    // ---- workspace layout (256B aligned slabs) ----
    char*  ws  = (char*)d_ws;
    size_t off = 0;
    auto take = [&](size_t bytes) -> void* {
        void* p = ws + off;
        off += (bytes + 255) & ~(size_t)255;
        return p;
    };
    float*    h32    = (float*)   take((size_t)ROWS * D_  * 4);
    float*    y32    = (float*)   take((size_t)ROWS * D_  * 4);
    float*    attn32 = (float*)   take((size_t)ROWS * D_  * 4);
    float*    z32    = (float*)   take((size_t)ROWS * D_  * 4);
    _Float16* h16    = (_Float16*)take((size_t)ROWS * D_  * 2);
    _Float16* y16    = (_Float16*)take((size_t)ROWS * D_  * 2);
    _Float16* msg16  = (_Float16*)take((size_t)ROWS * D_  * 2);
    _Float16* zact16 = (_Float16*)take((size_t)ROWS * HID_ * 2);
    _Float16* wqT    = (_Float16*)take((size_t)D_   * D_   * 2);
    _Float16* wkvT   = (_Float16*)take((size_t)D_   * 2*D_ * 2);
    _Float16* woT    = (_Float16*)take((size_t)D_   * D_   * 2);
    _Float16* wgT    = (_Float16*)take((size_t)D_   * HID_ * 2);
    _Float16* wvT    = (_Float16*)take((size_t)D_   * HID_ * 2);
    _Float16* wpT    = (_Float16*)take((size_t)HID_ * D_   * 2);
    // Overlay: {q32,kv32} (attention phase) shares memory with {g32,v32}
    // (FFN phase) -- lifetimes are disjoint.
    char*  big  = (char*)take((size_t)ROWS * HID_ * 4 * 2);
    float* q32  = (float*)big;
    float* kv32 = (float*)(big + (size_t)ROWS * D_   * 4);
    float* g32  = (float*)big;
    float* vv32 = (float*)(big + (size_t)ROWS * HID_ * 4);

    const dim3 blk256(256);

    // ---- weights -> f16 transposed (Bt = N x K) ----
    auto launchT = [&](const float* w, _Float16* wt, int K, int N) {
        size_t n = (size_t)K * N;
        transpose_f16_kernel<<<dim3((unsigned)((n + 255) / 256)), blk256, 0, stream>>>(w, wt, K, N);
    };
    launchT(wq,     wqT,  D_,   D_);
    launchT(wkv,    wkvT, D_,   2 * D_);
    launchT(wo,     woT,  D_,   D_);
    launchT(w_gate, wgT,  D_,   HID_);
    launchT(w_val,  wvT,  D_,   HID_);
    launchT(w_proj, wpT,  HID_, D_);

    auto gemm = [&](const _Float16* A, const _Float16* Bt, float* C, int N, int K) {
        gemm_f16_wmma<<<dim3(ROWS / 128, N / 64), blk256, 0, stream>>>(A, Bt, C, ROWS, N, K);
    };

    // 1) h = LN(x) ; also h16
    ln_kernel<<<ROWS, blk256, 0, stream>>>(x, nullptr, pre_g, pre_b,
                                           h32, h16, nullptr, nullptr, nullptr);
    // 2) q = h@wq, kv = h@wkv  (WMMA)
    gemm(h16, wqT,  q32,  D_,     D_);
    gemm(h16, wkvT, kv32, 2 * D_, D_);
    // 3) sliding-window sigmoid attention -> msg16
    attn_kernel<<<ROWS * H_, dim3(64), 0, stream>>>(q32, kv32, msg16);
    // 4) attnout = msg@wo  (WMMA)
    gemm(msg16, woT, attn32, D_, D_);
    // 5) y = LN(h + attnout) * (1 + amp*sin(t*freqs + phases)) ; also y16
    ln_kernel<<<ROWS, blk256, 0, stream>>>(h32, attn32, attn_g, attn_b,
                                           y32, y16, freqs, phases, amp);
    // 6) gate/val GEMMs  (WMMA)
    gemm(y16, wgT, g32,  HID_, D_);
    gemm(y16, wvT, vv32, HID_, D_);
    // 7) zact = silu(gate) * val -> f16
    {
        size_t n = (size_t)ROWS * HID_;
        silu_mul_kernel<<<dim3((unsigned)((n + 255) / 256)), blk256, 0, stream>>>(g32, vv32, zact16, n);
    }
    // 8) z = zact @ w_proj  (WMMA, K=1536)
    gemm(zact16, wpT, z32, D_, HID_);
    // 9) out = LN(y + z)
    ln_kernel<<<ROWS, blk256, 0, stream>>>(y32, z32, ffn_g, ffn_b,
                                           (float*)d_out, nullptr, nullptr, nullptr, nullptr);
}
